// GPSA_42502996361548
// MI455X (gfx1250) — compile-verified, matrix operand order checked by reading, not run on hardware
//
#include <hip/hip_runtime.h>
#include <hip/hip_bf16.h>
#include <cstdint>

typedef __attribute__((ext_vector_type(16))) __bf16 v16bf;
typedef __attribute__((ext_vector_type(8)))  float  v8f;

#define DIMC   768
#define HEADS  16
#define HD     48
#define HDP    64          // head dim padded to WMMA K granularity
#define NTOK   576         // tokens per image (24x24)
#define BATCH  8
#define MTOT   (BATCH*NTOK)        // 4608 rows
#define NQKV   (2*DIMC + DIMC)     // 2304 qkv output features
#define SCALE  0.14433756729740643f  // 48^-0.5

// -------- WMMA helpers (CDNA5 16x16x32 bf16, fp32 accum) --------------------
__device__ __forceinline__ v8f wmma_bf16(v16bf a, v16bf b, v8f c) {
  return __builtin_amdgcn_wmma_f32_16x16x32_bf16(false, a, false, b, (short)0, c,
                                                 false, false);
}

// Per-lane fragment load. rowbase already points at this lane's row (A) or
// column-row (B); K elements are contiguous. ISA 16-bit layout:
// element e -> K = k0 + (e>>3)*16 + (lane>=16 ? 8 : 0) + (e&7)
// (compiler proves pairs contiguous -> emits global_load_b128 pairs)
__device__ __forceinline__ v16bf load_frag16(const __bf16* rowbase, int half, int k0) {
  v16bf f;
#pragma unroll
  for (int e = 0; e < 16; ++e) {
    int k = k0 + ((e >> 3) << 4) + (half << 3) + (e & 7);
    f[e] = rowbase[k];
  }
  return f;
}

// ---------------- K0: fp32 -> bf16 packing + zero padding -------------------
__global__ void k0_pack(const float* __restrict__ x,
                        const float* __restrict__ w_qk,
                        const float* __restrict__ w_v,
                        const float* __restrict__ w_proj,
                        __bf16* __restrict__ xb,
                        __bf16* __restrict__ wqkvb,
                        __bf16* __restrict__ wpb,
                        __bf16* __restrict__ qarr,
                        __bf16* __restrict__ karr) {
  const long NX  = (long)MTOT * DIMC;
  const long NQK = 2L * DIMC * DIMC;
  const long NV  = (long)DIMC * DIMC;
  const long NP  = (long)DIMC * DIMC;
  const long NZ  = (long)BATCH * HEADS * NTOK * HDP;
  const long total = NX + NQK + NV + NP + 2 * NZ;
  for (long i = (long)blockIdx.x * blockDim.x + threadIdx.x; i < total;
       i += (long)gridDim.x * blockDim.x) {
    if (i < NX) {
      xb[i] = (__bf16)x[i];
    } else if (i < NX + NQK) {
      wqkvb[i - NX] = (__bf16)w_qk[i - NX];
    } else if (i < NX + NQK + NV) {
      wqkvb[i - NX] = (__bf16)w_v[i - NX - NQK];   // appended after w_qk rows
    } else if (i < NX + NQK + NV + NP) {
      long j = i - NX - NQK - NV;
      wpb[j] = (__bf16)w_proj[j];
    } else {
      long j = i - (NX + NQK + NV + NP);
      if (j < NZ) qarr[j] = (__bf16)0.0f;
      else        karr[j - NZ] = (__bf16)0.0f;
    }
  }
}

// scatter one 16x16 fp32 C tile into q / k / vT per output-feature routing
__device__ __forceinline__ void scatter_qkv(v8f acc, int mrow0, int f, int half,
                                            __bf16* __restrict__ qarr,
                                            __bf16* __restrict__ karr,
                                            __bf16* __restrict__ vT) {
#pragma unroll
  for (int j = 0; j < 8; ++j) {
    int gm = mrow0 + j + 8 * half;            // global token row
    int bi = gm / NTOK, n = gm % NTOK;
    float val = acc[j];
    if (f < DIMC) {                           // q (pre-scale)
      int h = f / HD, d = f % HD;
      qarr[(((long)bi * HEADS + h) * NTOK + n) * HDP + d] = (__bf16)(val * SCALE);
    } else if (f < 2 * DIMC) {                // k
      int f2 = f - DIMC; int h = f2 / HD, d = f2 % HD;
      karr[(((long)bi * HEADS + h) * NTOK + n) * HDP + d] = (__bf16)val;
    } else {                                  // v, transposed [B,H,48,576]
      int f2 = f - 2 * DIMC; int h = f2 / HD, d = f2 % HD;
      vT[(((long)bi * HEADS + h) * HD + d) * NTOK + n] = (__bf16)val;
    }
  }
}

// ---------------- K1: qkv projection GEMM (WMMA, 2x2 reg blocking) ----------
// Y[4608,2304] = xb[4608,768] * wqkvb[2304,768]^T ; wave owns a 32x32 tile.
__global__ void __launch_bounds__(128)
k1_gemm_qkv(const __bf16* __restrict__ xb, const __bf16* __restrict__ wqkvb,
            __bf16* __restrict__ qarr, __bf16* __restrict__ karr,
            __bf16* __restrict__ vT) {
  const int wave = (blockIdx.x * blockDim.x + threadIdx.x) >> 5;   // 10368 waves
  const int l = threadIdx.x & 31, half = l >> 4, r = l & 15;
  const int tm = wave % (MTOT / 32);          // 144 row blocks (32 rows each)
  const int tn = wave / (MTOT / 32);          // 72 col blocks (32 cols each)
  const int m0 = tm * 32, n0 = tn * 32;

  const __bf16* a0 = xb    + (long)(m0 + r) * DIMC;
  const __bf16* a1 = xb    + (long)(m0 + 16 + r) * DIMC;
  const __bf16* b0 = wqkvb + (long)(n0 + r) * DIMC;
  const __bf16* b1 = wqkvb + (long)(n0 + 16 + r) * DIMC;

  v8f acc00 = {}, acc01 = {}, acc10 = {}, acc11 = {};
#pragma unroll 2
  for (int kb = 0; kb < DIMC; kb += 32) {
    v16bf fa0 = load_frag16(a0, half, kb);
    v16bf fa1 = load_frag16(a1, half, kb);
    v16bf fb0 = load_frag16(b0, half, kb);
    v16bf fb1 = load_frag16(b1, half, kb);
    acc00 = wmma_bf16(fa0, fb0, acc00);
    acc01 = wmma_bf16(fa0, fb1, acc01);
    acc10 = wmma_bf16(fa1, fb0, acc10);
    acc11 = wmma_bf16(fa1, fb1, acc11);
  }
  scatter_qkv(acc00, m0,      n0 + r,      half, qarr, karr, vT);
  scatter_qkv(acc01, m0,      n0 + 16 + r, half, qarr, karr, vT);
  scatter_qkv(acc10, m0 + 16, n0 + r,      half, qarr, karr, vT);
  scatter_qkv(acc11, m0 + 16, n0 + 16 + r, half, qarr, karr, vT);
}

// ---------------- K2: positional softmax [H,576,576] ------------------------
__global__ void __launch_bounds__(256)
k2_pos(const float* __restrict__ w_pos, const float* __restrict__ b_pos,
       float* __restrict__ pos) {
  __shared__ float red[256];
  const int h = blockIdx.x / NTOK;
  const int n = blockIdx.x % NTOK;
  const float w0 = w_pos[h * 3 + 0], w1 = w_pos[h * 3 + 1],
              w2 = w_pos[h * 3 + 2], bp = b_pos[h];
  const int nx = n % 24, ny = n / 24;
  const int t = threadIdx.x;

  float mx = -1e30f;
  for (int m = t; m < NTOK; m += 256) {
    float dx = (float)(m % 24 - nx), dy = (float)(m / 24 - ny);
    float lg = w0 * dx + w1 * dy + w2 * (dx * dx + dy * dy) + bp;
    mx = fmaxf(mx, lg);
  }
  red[t] = mx; __syncthreads();
  for (int s = 128; s > 0; s >>= 1) {
    if (t < s) red[t] = fmaxf(red[t], red[t + s]);
    __syncthreads();
  }
  mx = red[0]; __syncthreads();

  float sum = 0.f;
  for (int m = t; m < NTOK; m += 256) {
    float dx = (float)(m % 24 - nx), dy = (float)(m / 24 - ny);
    float lg = w0 * dx + w1 * dy + w2 * (dx * dx + dy * dy) + bp;
    sum += __expf(lg - mx);
  }
  red[t] = sum; __syncthreads();
  for (int s = 128; s > 0; s >>= 1) {
    if (t < s) red[t] += red[t + s];
    __syncthreads();
  }
  const float inv = 1.0f / red[0];

  float* prow = pos + ((long)h * NTOK + n) * NTOK;
  for (int m = t; m < NTOK; m += 256) {
    float dx = (float)(m % 24 - nx), dy = (float)(m / 24 - ny);
    float lg = w0 * dx + w1 * dy + w2 * (dx * dx + dy * dy) + bp;
    prow[m] = __expf(lg - mx) * inv;
  }
}

// ---------------- K3: gated attention (WMMA x2 + parallel softmax) ----------
// One block = (b, h, 16-query stripe); 4 waves.
__global__ void __launch_bounds__(128)
k3_attn(const __bf16* __restrict__ qarr, const __bf16* __restrict__ karr,
        const __bf16* __restrict__ vT, const float* __restrict__ pos,
        const float* __restrict__ gating, __bf16* __restrict__ ctx) {
  __shared__ float  Sf[16][NTOK];   // raw scores, 36 KB
  __shared__ __bf16 At[16][NTOK];   // blended attn, 18 KB
  __shared__ float  red16[16][9];   // softmax partials (pad col to 9)

  const int blk = blockIdx.x;                 // B*H*36 = 4608
  const int rt = blk % (NTOK / 16);
  const int bh = blk / (NTOK / 16);
  const int h  = bh % HEADS, bi = bh / HEADS;
  const int m0 = rt * 16;
  const int wid = threadIdx.x >> 5;
  const int l = threadIdx.x & 31, half = l >> 4, r = l & 15;

  // q fragments for this 16-row stripe (K = 64, padded)
  const __bf16* qb = qarr + (((long)bi * HEADS + h) * NTOK + m0) * HDP + (long)r * HDP;
  v16bf qa0 = load_frag16(qb, half, 0);
  v16bf qa1 = load_frag16(qb, half, 32);

  // Phase 1: S[16,576] = q * k^T  (each wave owns 9 column tiles)
  for (int ci = 0; ci < 9; ++ci) {
    int c0 = (wid * 9 + ci) * 16;
    const __bf16* kb = karr + (((long)bi * HEADS + h) * NTOK + c0) * HDP + (long)r * HDP;
    v16bf b0 = load_frag16(kb, half, 0);
    v16bf b1 = load_frag16(kb, half, 32);
    v8f acc = {};
    acc = wmma_bf16(qa0, b0, acc);
    acc = wmma_bf16(qa1, b1, acc);
#pragma unroll
    for (int j = 0; j < 8; ++j) Sf[j + 8 * half][c0 + r] = acc[j];
  }
  __syncthreads();

  // Phase 2: row softmax + gated blend, all 128 threads.
  // thread -> (row = tid&15, segment = tid>>4 of 72 columns)
  {
    const int row = threadIdx.x & 15;
    const int seg = threadIdx.x >> 4;         // 0..7
    const int mlo = seg * 72, mhi = mlo + 72;

    float mx = -1e30f;
    for (int m = mlo; m < mhi; ++m) mx = fmaxf(mx, Sf[row][m]);
    red16[row][seg] = mx;
    __syncthreads();
    float rmx = -1e30f;
#pragma unroll
    for (int s = 0; s < 8; ++s) rmx = fmaxf(rmx, red16[row][s]);

    float sum = 0.f;
    for (int m = mlo; m < mhi; ++m) sum += __expf(Sf[row][m] - rmx);
    __syncthreads();                          // all reads of red16 done
    red16[row][seg] = sum;
    __syncthreads();
    float rsum = 0.f;
#pragma unroll
    for (int s = 0; s < 8; ++s) rsum += red16[row][s];

    const float inv = 1.0f / rsum;
    const float g = 1.0f / (1.0f + __expf(-gating[h]));
    const float* pr = pos + ((long)h * NTOK + (m0 + row)) * NTOK;
    for (int m = mlo; m < mhi; ++m) {
      float a = (1.0f - g) * __expf(Sf[row][m] - rmx) * inv + g * pr[m];
      At[row][m] = (__bf16)a;   // blend sums to 1 -> renorm is identity
    }
  }
  __syncthreads();

  // Phase 3: O[16,48] = attn[16,576] * v[576,48]  (waves 0..2, one d-tile each)
  if (wid < 3) {
    const __bf16* vb = vT + (((long)bi * HEADS + h) * HD + wid * 16) * NTOK + (long)r * NTOK;
    v8f acc = {};
    for (int kb2 = 0; kb2 < NTOK; kb2 += 32) {
      v16bf a  = load_frag16(&At[r][0], half, kb2);  // LDS (ds_load path)
      v16bf bv = load_frag16(vb, half, kb2);
      acc = wmma_bf16(a, bv, acc);
    }
#pragma unroll
    for (int j = 0; j < 8; ++j) {
      long tok = (long)bi * NTOK + m0 + j + 8 * half;
      ctx[tok * DIMC + h * HD + wid * 16 + r] = (__bf16)acc[j];
    }
  }
}

// ---------------- K4: output projection GEMM + bias (2x2 reg blocking) ------
__global__ void __launch_bounds__(128)
k4_proj(const __bf16* __restrict__ ctx, const __bf16* __restrict__ wpb,
        const float* __restrict__ bias, float* __restrict__ out) {
  const int wave = (blockIdx.x * blockDim.x + threadIdx.x) >> 5;   // 3456 waves
  const int l = threadIdx.x & 31, half = l >> 4, r = l & 15;
  const int tm = wave % (MTOT / 32);          // 144 row blocks
  const int tn = wave / (MTOT / 32);          // 24 col blocks
  const int m0 = tm * 32, n0 = tn * 32;

  const __bf16* a0 = ctx + (long)(m0 + r) * DIMC;
  const __bf16* a1 = ctx + (long)(m0 + 16 + r) * DIMC;
  const __bf16* b0 = wpb + (long)(n0 + r) * DIMC;
  const __bf16* b1 = wpb + (long)(n0 + 16 + r) * DIMC;

  v8f acc00 = {}, acc01 = {}, acc10 = {}, acc11 = {};
#pragma unroll 2
  for (int kb = 0; kb < DIMC; kb += 32) {
    v16bf fa0 = load_frag16(a0, half, kb);
    v16bf fa1 = load_frag16(a1, half, kb);
    v16bf fb0 = load_frag16(b0, half, kb);
    v16bf fb1 = load_frag16(b1, half, kb);
    acc00 = wmma_bf16(fa0, fb0, acc00);
    acc01 = wmma_bf16(fa0, fb1, acc01);
    acc10 = wmma_bf16(fa1, fb0, acc10);
    acc11 = wmma_bf16(fa1, fb1, acc11);
  }

  const float bv0 = bias[n0 + r];
  const float bv1 = bias[n0 + 16 + r];
#pragma unroll
  for (int j = 0; j < 8; ++j) {
    long gm0 = m0 + j + 8 * half;
    long gm1 = m0 + 16 + j + 8 * half;
    out[gm0 * DIMC + n0 + r]      = acc00[j] + bv0;
    out[gm0 * DIMC + n0 + 16 + r] = acc01[j] + bv1;
    out[gm1 * DIMC + n0 + r]      = acc10[j] + bv0;
    out[gm1 * DIMC + n0 + 16 + r] = acc11[j] + bv1;
  }
}

// ---------------- host launcher ---------------------------------------------
extern "C" void kernel_launch(void* const* d_in, const int* in_sizes, int n_in,
                              void* d_out, int out_size, void* d_ws, size_t ws_size,
                              hipStream_t stream) {
  (void)in_sizes; (void)n_in; (void)out_size; (void)ws_size;
  const float* x      = (const float*)d_in[0];
  const float* w_qk   = (const float*)d_in[1];
  const float* w_v    = (const float*)d_in[2];
  const float* w_proj = (const float*)d_in[3];
  const float* b_proj = (const float*)d_in[4];
  const float* w_pos  = (const float*)d_in[5];
  const float* b_pos  = (const float*)d_in[6];
  const float* gating = (const float*)d_in[7];
  float* out = (float*)d_out;

  size_t off = 0;
  auto carve = [&](size_t bytes) -> void* {
    void* p = (char*)d_ws + off;
    off = (off + bytes + 255) & ~(size_t)255;
    return p;
  };
  __bf16* xb    = (__bf16*)carve((size_t)MTOT * DIMC * 2);               // 7.08 MB
  __bf16* wqkvb = (__bf16*)carve((size_t)NQKV * DIMC * 2);               // 3.54 MB
  __bf16* wpb   = (__bf16*)carve((size_t)DIMC * DIMC * 2);               // 1.18 MB
  __bf16* qarr  = (__bf16*)carve((size_t)BATCH * HEADS * NTOK * HDP * 2);// 9.44 MB
  __bf16* karr  = (__bf16*)carve((size_t)BATCH * HEADS * NTOK * HDP * 2);// 9.44 MB
  __bf16* vTm   = (__bf16*)carve((size_t)BATCH * HEADS * HD * NTOK * 2); // 7.08 MB
  float*  pos   = (float*)carve((size_t)HEADS * NTOK * NTOK * 4);        // 21.2 MB
  __bf16* ctx   = (__bf16*)carve((size_t)MTOT * DIMC * 2);               // 7.08 MB

  k0_pack<<<1024, 256, 0, stream>>>(x, w_qk, w_v, w_proj, xb, wqkvb, wpb, qarr, karr);

  // 144*72 = 10368 waves, 4 waves/block -> 2592 blocks
  k1_gemm_qkv<<<2592, 128, 0, stream>>>(xb, wqkvb, qarr, karr, vTm);

  k2_pos<<<HEADS * NTOK, 256, 0, stream>>>(w_pos, b_pos, pos);

  k3_attn<<<BATCH * HEADS * (NTOK / 16), 128, 0, stream>>>(qarr, karr, vTm, pos,
                                                           gating, ctx);

  // 144*24 = 3456 waves -> 864 blocks
  k4_proj<<<864, 128, 0, stream>>>(ctx, wpb, b_proj, out);
}
